// LinkedAttention_25924422599209
// MI455X (gfx1250) — matrix operand order repaired and checked by reading, not compiled, not run on hardware
//
#include <hip/hip_runtime.h>
#include <hip/hip_bf16.h>

typedef __attribute__((ext_vector_type(16))) __bf16 v16bf;
typedef __attribute__((ext_vector_type(8)))  __bf16 v8bf;
typedef __attribute__((ext_vector_type(8)))  float  v8f;

#define B_   32
#define T_   2048
#define D_   1024
#define DIM_ 1024

// ---------------------------------------------------------------------------
// Repack Wy (f32, row-major D x DIM) into bf16 WMMA B-fragment order:
// Wy_sw[((ntile*32 + kstep)*32 + lane)*16 + i]
//   lane<16 : col = ntile*16+lane,    K = kstep*32 + i        (i=0..15)
//   lane>=16: col = ntile*16+lane-16, K = kstep*32 + 16 + i
// so each lane's 16 B-matrix elements are one aligned 32-byte load.
// ---------------------------------------------------------------------------
__global__ __launch_bounds__(256) void la_swizzle_wy(const float* __restrict__ Wy,
                                                     __bf16* __restrict__ Wy_sw) {
    size_t o = (size_t)blockIdx.x * 256 + threadIdx.x;   // 0 .. 1024*1024-1
    int i    = (int)(o & 15);
    int lane = (int)((o >> 4) & 31);
    int ks   = (int)((o >> 9) & 31);
    int nt   = (int)(o >> 14);
    int col  = nt * 16 + (lane & 15);
    int k    = ks * 32 + (lane >> 4) * 16 + i;
    Wy_sw[o] = (__bf16)Wy[(size_t)k * DIM_ + col];
}

// ---------------------------------------------------------------------------
// hproj = last_state @ Wh ; tproj = last_state @ Wx      (B x DIM each)
// ---------------------------------------------------------------------------
__global__ __launch_bounds__(256) void la_proj(const float* __restrict__ last,
                                               const float* __restrict__ Wh,
                                               const float* __restrict__ Wx,
                                               float* __restrict__ hproj,
                                               float* __restrict__ tproj) {
    __shared__ float ls[D_];
    int b = blockIdx.x;
    int e = blockIdx.y * 256 + threadIdx.x;
    for (int d = threadIdx.x; d < D_; d += 256) ls[d] = last[(size_t)b * D_ + d];
    __syncthreads();
    float a1 = 0.f, a2 = 0.f;
    for (int d = 0; d < D_; ++d) {
        float l = ls[d];
        a1 = fmaf(l, Wh[(size_t)d * DIM_ + e], a1);
        a2 = fmaf(l, Wx[(size_t)d * DIM_ + e], a2);
    }
    hproj[(size_t)b * DIM_ + e] = a1;
    tproj[(size_t)b * DIM_ + e] = a2;
}

// ---------------------------------------------------------------------------
// Fused scores kernel: scores[m] = sum_e relu((x_row @ Wy)[e] + hproj[b,e]) * w[e]
// One workgroup (8 waves) per 16-row M-tile; bf16 WMMA 16x16x32 with explicit
// ping-pong B-fragment buffers (no rotation copies): each half-step loads
// directly into the buffer consumed one half-step later.
// ---------------------------------------------------------------------------
__global__ __launch_bounds__(256) void la_scores(const float* __restrict__ x,
                                                 const __bf16* __restrict__ Wy_sw,
                                                 const float* __restrict__ hproj,
                                                 const float* __restrict__ w,
                                                 float* __restrict__ scores) {
    __shared__ __align__(32) __bf16 sA[16 * D_];     // A tile, bf16, row-major
    __shared__ float sPart[8][16];

    const int tid   = threadIdx.x;
    const int mtile = blockIdx.x;                    // 0..4095

    // --- stage A tile (16 contiguous rows of input_states) to LDS as bf16 ---
    const float4* src = (const float4*)(x + (size_t)mtile * 16 * D_);
    for (int i = tid; i < 16 * D_ / 4; i += 256) {
        float4 v = src[i];
        int o = i * 4;
        sA[o + 0] = (__bf16)v.x;
        sA[o + 1] = (__bf16)v.y;
        sA[o + 2] = (__bf16)v.z;
        sA[o + 3] = (__bf16)v.w;
    }
    __syncthreads();

    const int wave = tid >> 5;
    const int lane = tid & 31;
    const int row  = lane & 15;
    const int half = lane >> 4;                      // 0 or 1
    const __bf16* sArow = sA + row * D_ + half * 16; // A-frag layout base
    const int nbase = wave * 8;                      // 8 N-tiles per wave

    // Per-lane base of this wave's B fragments
    const __bf16* pbBase = Wy_sw + (((size_t)nbase * 32) * 32 + lane) * 16;
    const size_t  nStride = (size_t)32 * 32 * 16;    // 16384 elems per n-tile
    const size_t  kStride = (size_t)32 * 16;         // 512 elems per k-step

    v8f C[8] = {};
    v16bf ba[8], bb[8];

#define LA_LOAD_B(buf, kstep)                                                  \
    {                                                                          \
        const __bf16* p_ = pbBase + (size_t)(kstep) * kStride;                 \
        _Pragma("unroll")                                                      \
        for (int n_ = 0; n_ < 8; ++n_)                                         \
            buf[n_] = *(const v16bf*)(p_ + (size_t)n_ * nStride);              \
    }

#define LA_WMMA(buf, kstep)                                                    \
    {                                                                          \
        const __bf16* pa_ = sArow + (kstep) * 32;                              \
        v8bf alo_ = *(const v8bf*)(pa_);                                       \
        v8bf ahi_ = *(const v8bf*)(pa_ + 8);                                   \
        v16bf af_ = __builtin_shufflevector(alo_, ahi_,                        \
                     0, 1, 2, 3, 4, 5, 6, 7, 8, 9, 10, 11, 12, 13, 14, 15);   \
        _Pragma("unroll")                                                      \
        for (int n_ = 0; n_ < 8; ++n_)                                         \
            C[n_] = __builtin_amdgcn_wmma_f32_16x16x32_bf16(                   \
                        false, af_, false, buf[n_], (short)0, C[n_],           \
                        false, false);                                         \
    }

    LA_LOAD_B(ba, 0);
    for (int ks = 0; ks < 30; ks += 2) {
        LA_LOAD_B(bb, ks + 1);       // prefetch odd step
        LA_WMMA(ba, ks);             // consume even step
        LA_LOAD_B(ba, ks + 2);       // prefetch next even step
        LA_WMMA(bb, ks + 1);         // consume odd step
    }
    LA_LOAD_B(bb, 31);
    LA_WMMA(ba, 30);
    LA_WMMA(bb, 31);

#undef LA_LOAD_B
#undef LA_WMMA

    // --- epilogue: + hproj, relu, * w, reduce over columns ---
    const int bidx = mtile >> 7;                     // 2048 rows per batch
    const int col  = lane & 15;
    float acc[8] = {0.f, 0.f, 0.f, 0.f, 0.f, 0.f, 0.f, 0.f};
#pragma unroll
    for (int n = 0; n < 8; ++n) {
        int nt   = nbase + n;
        float hp = hproj[(size_t)bidx * DIM_ + nt * 16 + col];
        float wv = w[nt * 16 + col];
#pragma unroll
        for (int v = 0; v < 8; ++v) {
            float m = C[n][v] + hp;
            m = m > 0.f ? m : 0.f;
            acc[v] += m * wv;                        // row = half*8 + v
        }
    }
#pragma unroll
    for (int v = 0; v < 8; ++v) {                    // reduce 16 columns/half
        float s = acc[v];
        s += __shfl_xor(s, 1);
        s += __shfl_xor(s, 2);
        s += __shfl_xor(s, 4);
        s += __shfl_xor(s, 8);
        acc[v] = s;
    }
    if ((lane & 15) == 0) {
#pragma unroll
        for (int v = 0; v < 8; ++v) sPart[wave][half * 8 + v] = acc[v];
    }
    __syncthreads();
    if (tid < 16) {
        float s = 0.f;
        for (int wv2 = 0; wv2 < 8; ++wv2) s += sPart[wv2][tid];
        scores[(size_t)mtile * 16 + tid] = s;
    }
}

// ---------------------------------------------------------------------------
// Softmax over T per batch row (in-place on scores)
// ---------------------------------------------------------------------------
__global__ __launch_bounds__(256) void la_softmax(float* __restrict__ scores) {
    __shared__ float red[256];
    int b = blockIdx.x, tid = threadIdx.x;
    float* s = scores + (size_t)b * T_;
    float mx = -3.402823466e38f;
    for (int t = tid; t < T_; t += 256) mx = fmaxf(mx, s[t]);
    red[tid] = mx; __syncthreads();
    for (int off = 128; off > 0; off >>= 1) {
        if (tid < off) red[tid] = fmaxf(red[tid], red[tid + off]);
        __syncthreads();
    }
    mx = red[0]; __syncthreads();
    float sum = 0.f;
    for (int t = tid; t < T_; t += 256) { float e = __expf(s[t] - mx); s[t] = e; sum += e; }
    red[tid] = sum; __syncthreads();
    for (int off = 128; off > 0; off >>= 1) {
        if (tid < off) red[tid] += red[tid + off];
        __syncthreads();
    }
    float inv = 1.f / red[0];
    __syncthreads();
    for (int t = tid; t < T_; t += 256) s[t] *= inv;
}

// ---------------------------------------------------------------------------
// r[b,d] = sum_t alpha[b,t] * x[b,t,d] : deterministic chunked partials
// ---------------------------------------------------------------------------
__global__ __launch_bounds__(256) void la_rpart(const float* __restrict__ x,
                                                const float* __restrict__ alpha,
                                                float* __restrict__ rpart) {
    int b = blockIdx.x, ch = blockIdx.y, tid = threadIdx.x;
    float a0 = 0.f, a1 = 0.f, a2 = 0.f, a3 = 0.f;
    const float* xb = x + ((size_t)b * T_ + (size_t)ch * 128) * D_;
    const float* al = alpha + (size_t)b * T_ + (size_t)ch * 128;
    for (int t = 0; t < 128; ++t) {
        float a = al[t];
        const float* xr = xb + (size_t)t * D_;
        a0 = fmaf(a, xr[tid], a0);
        a1 = fmaf(a, xr[tid + 256], a1);
        a2 = fmaf(a, xr[tid + 512], a2);
        a3 = fmaf(a, xr[tid + 768], a3);
    }
    float* rp = rpart + ((size_t)b * 16 + ch) * D_;
    rp[tid] = a0; rp[tid + 256] = a1; rp[tid + 512] = a2; rp[tid + 768] = a3;
}

__global__ __launch_bounds__(256) void la_rred(const float* __restrict__ rpart,
                                               float* __restrict__ r) {
    int b = blockIdx.x;
    int d = blockIdx.y * 256 + threadIdx.x;
    float s = 0.f;
    for (int c = 0; c < 16; ++c) s += rpart[((size_t)b * 16 + c) * D_ + d];
    r[(size_t)b * D_ + d] = s;
}

// ---------------------------------------------------------------------------
// out = relu(r @ Wp + tproj)
// ---------------------------------------------------------------------------
__global__ __launch_bounds__(256) void la_out(const float* __restrict__ r,
                                              const float* __restrict__ Wp,
                                              const float* __restrict__ tproj,
                                              float* __restrict__ out) {
    __shared__ float rs[D_];
    int b = blockIdx.x;
    int e = blockIdx.y * 256 + threadIdx.x;
    for (int d = threadIdx.x; d < D_; d += 256) rs[d] = r[(size_t)b * D_ + d];
    __syncthreads();
    float a = 0.f;
    for (int d = 0; d < D_; ++d) a = fmaf(rs[d], Wp[(size_t)d * DIM_ + e], a);
    a += tproj[(size_t)b * DIM_ + e];
    out[(size_t)b * DIM_ + e] = a > 0.f ? a : 0.f;
}

// ---------------------------------------------------------------------------
extern "C" void kernel_launch(void* const* d_in, const int* in_sizes, int n_in,
                              void* d_out, int out_size, void* d_ws, size_t ws_size,
                              hipStream_t stream) {
    const float* x    = (const float*)d_in[0];   // (B,T,D)
    const float* last = (const float*)d_in[1];   // (B,D)
    const float* Wy   = (const float*)d_in[2];   // (D,DIM)
    const float* Wh   = (const float*)d_in[3];   // (D,DIM)
    const float* w    = (const float*)d_in[4];   // (DIM,1)
    const float* Wp   = (const float*)d_in[5];   // (D,DIM)
    const float* Wx   = (const float*)d_in[6];   // (D,DIM)
    float* out        = (float*)d_out;           // (B,DIM)

    char* ws = (char*)d_ws;
    float*  hproj  = (float*)(ws);                  // 128 KB
    float*  tproj  = (float*)(ws + (128 << 10));    // 128 KB
    float*  scores = (float*)(ws + (256 << 10));    // 256 KB (becomes alpha)
    float*  r      = (float*)(ws + (512 << 10));    // 128 KB
    float*  rpart  = (float*)(ws + (640 << 10));    // 2 MB
    __bf16* Wy_sw  = (__bf16*)(ws + (2688 << 10));  // 2 MB

    la_swizzle_wy<<<4096, 256, 0, stream>>>(Wy, Wy_sw);
    la_proj<<<dim3(32, 4), 256, 0, stream>>>(last, Wh, Wx, hproj, tproj);
    la_scores<<<(B_ * T_) / 16, 256, 0, stream>>>(x, Wy_sw, hproj, w, scores);
    la_softmax<<<B_, 256, 0, stream>>>(scores);
    la_rpart<<<dim3(B_, 16), 256, 0, stream>>>(x, scores, rpart);
    la_rred<<<dim3(B_, 4), 256, 0, stream>>>(rpart, r);
    la_out<<<dim3(B_, 4), 256, 0, stream>>>(r, Wp, tproj, out);
}
